// GCN_17076789969171
// MI455X (gfx1250) — matrix operand order repaired and checked by reading, not compiled, not run on hardware
//
#include <hip/hip_runtime.h>
#include <stdint.h>

#define N_NODES 100000
#define N_EDGES 800000
#define F_IN    128
#define F_HID   512
#define F_OUT   40

typedef __attribute__((ext_vector_type(16))) __bf16 v16bf;
typedef __attribute__((ext_vector_type(8)))  float  v8f;

__device__ __forceinline__ __bf16 tobf(float f) {
    unsigned u = __builtin_bit_cast(unsigned, f);
    u = (u + 0x7FFFu + ((u >> 16) & 1u)) >> 16;   // RNE truncate f32 -> bf16
    return __builtin_bit_cast(__bf16, (unsigned short)u);
}

__device__ __forceinline__ uint32_t hash_u32(uint32_t h) {
    h *= 0x9E3779B9u;
    h ^= h >> 16; h *= 0x85EBCA6Bu; h ^= h >> 13; h *= 0xC2B2AE35u; h ^= h >> 16;
    return h;
}

// ---------------- utility kernels ----------------

__global__ __launch_bounds__(256) void zero_f32(float* __restrict__ p, int n) {
    int i = blockIdx.x * 256 + threadIdx.x;
    if (i < n) p[i] = 0.0f;
}

// deg[dst] += 1 for every edge (self-loop added in rsqrt kernel)
__global__ __launch_bounds__(256) void deg_kernel(const int* __restrict__ ei,
                                                  float* __restrict__ deg) {
    int e = blockIdx.x * 256 + threadIdx.x;
    if (e < N_EDGES) unsafeAtomicAdd(&deg[ei[N_EDGES + e]], 1.0f);
}

// in-place: deg -> rsqrt(deg + 1)
__global__ __launch_bounds__(256) void dinv_kernel(float* __restrict__ deg) {
    int i = blockIdx.x * 256 + threadIdx.x;
    if (i < N_NODES) deg[i] = rsqrtf(deg[i] + 1.0f);
}

// ---------------- WMMA GEMM 1: h1p = dinv[row] * (X @ W1); agg1 seeded with h1p ----------------
// grid = (6250, 4), block = 256 (8 waves). Wave w handles N-tile blockIdx.y*8+w.
__global__ __launch_bounds__(256) void gemm1_kernel(const float* __restrict__ x,
                                                    const float* __restrict__ W1,
                                                    const float* __restrict__ dinv,
                                                    float* __restrict__ h1p,
                                                    float* __restrict__ agg1) {
    const int lane  = threadIdx.x & 31;
    const int wave  = threadIdx.x >> 5;
    const int half  = lane >> 4;          // which 16-lane half of the wave
    const int l16   = lane & 15;
    const int mBase = blockIdx.x * 16;
    const int nBase = (blockIdx.y * 8 + wave) * 16;

    v8f acc = {};
    for (int kk = 0; kk < F_IN; kk += 32) {
        // A tile 16x32 bf16: lane l16 = row M, per-lane ks follow ISA half-wave permute
        const float* arow = x + (size_t)(mBase + l16) * F_IN + kk;
        v16bf a, b;
#pragma unroll
        for (int v = 0; v < 8; ++v) {
            int k0 = (v < 4) ? (8 * half + 2 * v) : (16 + 8 * half + 2 * (v - 4));
            a[2 * v]     = tobf(arow[k0]);
            a[2 * v + 1] = tobf(arow[k0 + 1]);
        }
        // B tile 32x16 bf16: lane l16 = col N, VGPR j holds same K as A (half-wave permute), +1
#pragma unroll
        for (int j = 0; j < 8; ++j) {
            int k0 = kk + ((j < 4) ? (8 * half + 2 * j) : (16 + 8 * half + 2 * (j - 4)));
            b[2 * j]     = tobf(W1[(size_t)k0 * F_HID + nBase + l16]);
            b[2 * j + 1] = tobf(W1[(size_t)(k0 + 1) * F_HID + nBase + l16]);
        }
        acc = __builtin_amdgcn_wmma_f32_16x16x32_bf16(false, a, false, b,
                                                      (short)0, acc, false, false);
    }
    // D layout: VGPR r -> M = r + 8*half, N = l16.  Scale by dinv[row], seed agg (self-loop).
#pragma unroll
    for (int r = 0; r < 8; ++r) {
        int m = mBase + r + 8 * half;
        float val = acc[r] * dinv[m];
        size_t idx = (size_t)m * F_HID + nBase + l16;
        h1p[idx]  = val;
        agg1[idx] = val;
    }
}

// ---------------- edge scatter-add layer 1: agg1[dst] += h1p[src], 512 feats ----------------
// block = 256 (8 waves), 1 edge per wave, grid = 100000
__global__ __launch_bounds__(256) void agg1_kernel(const int* __restrict__ ei,
                                                   const float* __restrict__ h1p,
                                                   float* __restrict__ agg1) {
    int edge = blockIdx.x * 8 + (threadIdx.x >> 5);
    int lane = threadIdx.x & 31;
    int s = ei[edge];
    int d = ei[N_EDGES + edge];
    const float4* __restrict__ s4 = (const float4*)(h1p + (size_t)s * F_HID);
    float* __restrict__ drow = agg1 + (size_t)d * F_HID;
    __builtin_prefetch(s4, 0, 0);
#pragma unroll
    for (int it = 0; it < 4; ++it) {            // 32 lanes * 4 floats * 4 iters = 512
        int q = lane + 32 * it;
        float4 v = s4[q];
        float* dp = drow + 4 * q;
        unsafeAtomicAdd(dp + 0, v.x);
        unsafeAtomicAdd(dp + 1, v.y);
        unsafeAtomicAdd(dp + 2, v.z);
        unsafeAtomicAdd(dp + 3, v.w);
    }
}

// ---------------- epilogue 1: hidden = dropout(relu(dinv*agg1 + b1)), float4 ----------------
__global__ __launch_bounds__(256) void post1_kernel(const float* __restrict__ agg1,
                                                    const float* __restrict__ dinv,
                                                    const float* __restrict__ b1,
                                                    float* __restrict__ hidden) {
    int idx4 = blockIdx.x * 256 + threadIdx.x;    // over N*512/4
    int idx  = idx4 * 4;
    int i = idx >> 9;
    int f = idx & 511;                             // f % 4 == 0 -> same row, aligned
    float di = dinv[i];
    float4 a  = ((const float4*)agg1)[idx4];
    float4 bb = *(const float4*)(b1 + f);
    float4 r;
    r.x = fmaxf(di * a.x + bb.x, 0.0f);
    r.y = fmaxf(di * a.y + bb.y, 0.0f);
    r.z = fmaxf(di * a.z + bb.z, 0.0f);
    r.w = fmaxf(di * a.w + bb.w, 0.0f);
    r.x = (hash_u32((uint32_t)(idx + 0)) & 1u) ? 2.0f * r.x : 0.0f;
    r.y = (hash_u32((uint32_t)(idx + 1)) & 1u) ? 2.0f * r.y : 0.0f;
    r.z = (hash_u32((uint32_t)(idx + 2)) & 1u) ? 2.0f * r.z : 0.0f;
    r.w = (hash_u32((uint32_t)(idx + 3)) & 1u) ? 2.0f * r.w : 0.0f;
    ((float4*)hidden)[idx4] = r;
}

// ---------------- WMMA GEMM 2: h2p = dinv[row] * (hidden @ W2); agg2 seeded ----------------
// grid = 6250, block = 96 (3 waves = 3 N-tiles covering 40 cols, last tile masked)
__global__ __launch_bounds__(96) void gemm2_kernel(const float* __restrict__ hidden,
                                                   const float* __restrict__ W2,
                                                   const float* __restrict__ dinv,
                                                   float* __restrict__ h2p,
                                                   float* __restrict__ agg2) {
    const int lane  = threadIdx.x & 31;
    const int wave  = threadIdx.x >> 5;
    const int half  = lane >> 4;
    const int l16   = lane & 15;
    const int mBase = blockIdx.x * 16;
    const int nBase = wave * 16;
    const int ncol  = nBase + l16;
    const bool colOK = (ncol < F_OUT);

    v8f acc = {};
    for (int kk = 0; kk < F_HID; kk += 32) {
        const float* arow = hidden + (size_t)(mBase + l16) * F_HID + kk;
        v16bf a, b;
#pragma unroll
        for (int v = 0; v < 8; ++v) {
            int k0 = (v < 4) ? (8 * half + 2 * v) : (16 + 8 * half + 2 * (v - 4));
            a[2 * v]     = tobf(arow[k0]);
            a[2 * v + 1] = tobf(arow[k0 + 1]);
        }
#pragma unroll
        for (int j = 0; j < 8; ++j) {
            int k0 = kk + ((j < 4) ? (8 * half + 2 * j) : (16 + 8 * half + 2 * (j - 4)));
            b[2 * j]     = tobf(colOK ? W2[(size_t)k0 * F_OUT + ncol] : 0.0f);
            b[2 * j + 1] = tobf(colOK ? W2[(size_t)(k0 + 1) * F_OUT + ncol] : 0.0f);
        }
        acc = __builtin_amdgcn_wmma_f32_16x16x32_bf16(false, a, false, b,
                                                      (short)0, acc, false, false);
    }
#pragma unroll
    for (int r = 0; r < 8; ++r) {
        int m = mBase + r + 8 * half;
        float val = acc[r] * dinv[m];
        if (colOK) {
            size_t idx = (size_t)m * F_OUT + ncol;
            h2p[idx]  = val;
            agg2[idx] = val;
        }
    }
}

// ---------------- edge scatter-add layer 2: 40 feats/edge (10 float4 lanes) ----------------
__global__ __launch_bounds__(256) void agg2_kernel(const int* __restrict__ ei,
                                                   const float* __restrict__ h2p,
                                                   float* __restrict__ agg2) {
    int edge = blockIdx.x * 8 + (threadIdx.x >> 5);
    int lane = threadIdx.x & 31;
    int s = ei[edge];
    int d = ei[N_EDGES + edge];
    if (lane < F_OUT / 4) {
        float4 v = ((const float4*)(h2p + (size_t)s * F_OUT))[lane];
        float* dp = agg2 + (size_t)d * F_OUT + 4 * lane;
        unsafeAtomicAdd(dp + 0, v.x);
        unsafeAtomicAdd(dp + 1, v.y);
        unsafeAtomicAdd(dp + 2, v.z);
        unsafeAtomicAdd(dp + 3, v.w);
    }
}

// ---------------- epilogue 2: out = dinv*agg2 + b2, float4 ----------------
__global__ __launch_bounds__(256) void post2_kernel(const float* __restrict__ agg2,
                                                    const float* __restrict__ dinv,
                                                    const float* __restrict__ b2,
                                                    float* __restrict__ out) {
    int idx4 = blockIdx.x * 256 + threadIdx.x;     // over N*40/4 = 1,000,000
    if (idx4 >= N_NODES * F_OUT / 4) return;
    int idx = idx4 * 4;
    int i = idx / F_OUT;
    int f = idx % F_OUT;                            // f % 4 == 0 (40 % 4 == 0)
    float di = dinv[i];
    float4 a  = ((const float4*)agg2)[idx4];
    float4 bb = *(const float4*)(b2 + f);
    float4 r;
    r.x = di * a.x + bb.x;
    r.y = di * a.y + bb.y;
    r.z = di * a.z + bb.z;
    r.w = di * a.w + bb.w;
    ((float4*)out)[idx4] = r;
}

// ---------------- launcher ----------------

extern "C" void kernel_launch(void* const* d_in, const int* in_sizes, int n_in,
                              void* d_out, int out_size, void* d_ws, size_t ws_size,
                              hipStream_t stream) {
    const float* x  = (const float*)d_in[0];
    const int*   ei = (const int*)d_in[1];
    const float* W1 = (const float*)d_in[2];
    const float* b1 = (const float*)d_in[3];
    const float* W2 = (const float*)d_in[4];
    const float* b2 = (const float*)d_in[5];

    float* out    = (float*)d_out;                        // [N, 40]
    float* hidden = out + (size_t)N_NODES * F_OUT;        // [N, 512]

    float* dinv = (float*)d_ws;                           // N (deg -> rsqrt in place)
    float* h1p  = dinv + N_NODES;                         // N*512
    float* agg1 = h1p + (size_t)N_NODES * F_HID;          // N*512
    float* h2p  = agg1 + (size_t)N_NODES * F_HID;         // N*40
    float* agg2 = h2p + (size_t)N_NODES * F_OUT;          // N*40

    // degree + normalization
    zero_f32<<<(N_NODES + 255) / 256, 256, 0, stream>>>(dinv, N_NODES);
    deg_kernel<<<(N_EDGES + 255) / 256, 256, 0, stream>>>(ei, dinv);
    dinv_kernel<<<(N_NODES + 255) / 256, 256, 0, stream>>>(dinv);

    // layer 1
    gemm1_kernel<<<dim3(N_NODES / 16, F_HID / (16 * 8)), 256, 0, stream>>>(x, W1, dinv, h1p, agg1);
    agg1_kernel<<<N_EDGES / 8, 256, 0, stream>>>(ei, h1p, agg1);
    post1_kernel<<<(N_NODES * F_HID / 4) / 256, 256, 0, stream>>>(agg1, dinv, b1, hidden);

    // layer 2
    gemm2_kernel<<<N_NODES / 16, 96, 0, stream>>>(hidden, W2, dinv, h2p, agg2);
    agg2_kernel<<<N_EDGES / 8, 256, 0, stream>>>(ei, h2p, agg2);
    post2_kernel<<<(N_NODES * F_OUT / 4 + 255) / 256, 256, 0, stream>>>(agg2, dinv, b2, out);
}
